// MultiHeadSelfAttention_22711787061318
// MI455X (gfx1250) — matrix operand order repaired and checked by reading, not compiled, not run on hardware
//
#include <hip/hip_runtime.h>
#include <hip/hip_bf16.h>

// ---------------------------------------------------------------------------
// MHSA + RoPE for MI455X (gfx1250, wave32, WMMA f32_16x16x32_f16)
// B=4, N=2048, C=1024, H=16, D=64
//
// Workspace layout (halves), total ~75.5 MB (fits the 192 MB L2 together with
// the f32 inputs/outputs -> whole pipeline is L2-resident, compute-bound):
//   Xh   [8192,1024]   x in f16          (reused as attention output Oh)
//   Wqh  [3072,1024]   W_qkv f16 (row-major [out,in] == B^T for WMMA)
//   Woh  [1024,1024]   W_out f16
//   Qh   [B,H,N,D]     f16 (RoPE'd, pre-scaled by 1/sqrt(D))
//   Kh   [B,H,N,D]     f16 (RoPE'd)
//   Vt   [B,H,D,N]     f16 (transposed so B-fragments are contiguous)
// ---------------------------------------------------------------------------

typedef _Float16 v8h  __attribute__((ext_vector_type(8)));
typedef _Float16 v16h __attribute__((ext_vector_type(16)));
typedef float    v8f  __attribute__((ext_vector_type(8)));

// ---- WMMA fragment helpers (layouts per CDNA5 ISA §7.12.2) ------
// A (16x32 f16, M x K), A row-major [M,K]:
//   lane L: row m0 + (L&15); elements 0..7  -> K = k0 + 8*hi + e
//                            elements 8..15 -> K = k0 + 16 + 8*hi + (e-8)
__device__ __forceinline__ v16h load_a_frag(const _Float16* A, int lda, int m0, int k0) {
  const int lane = threadIdx.x & 31;
  const int hi = lane >> 4;
  const _Float16* p = A + (size_t)(m0 + (lane & 15)) * lda + k0 + 8 * hi;
  union { v16h v; v8h h[2]; } f;
  f.h[0] = *(const v8h*)(p);
  f.h[1] = *(const v8h*)(p + 16);
  return f.v;
}
// B (32x16 f16, K x N) loaded from B^T row-major [N,K]:
//   lane L: col n0 + (L&15); elements 0..7 -> K = k0 + 8*hi + e,
//                            elements 8..15 -> K = k0 + 16 + 8*hi + (e-8)
__device__ __forceinline__ v16h load_b_frag(const _Float16* Bt, int ldb, int n0, int k0) {
  const int lane = threadIdx.x & 31;
  const int hi = lane >> 4;
  const _Float16* p = Bt + (size_t)(n0 + (lane & 15)) * ldb + k0 + 8 * hi;
  union { v16h v; v8h h[2]; } f;
  f.h[0] = *(const v8h*)(p);
  f.h[1] = *(const v8h*)(p + 16);
  return f.v;
}
// C/D (16x16 f32): lane L holds col n0+(L&15), VGPR r holds row m0 + r + 8*hi.
__device__ __forceinline__ v8f wmma_f16(v16h a, v16h b, v8f c) {
  return __builtin_amdgcn_wmma_f32_16x16x32_f16(false, a, false, b, (short)0, c,
                                                false, false);
}

// ---------------------------------------------------------------------------
__global__ void cvt_f32_f16(const float* __restrict__ in, _Float16* __restrict__ out,
                            int n) {
  int i = blockIdx.x * blockDim.x + threadIdx.x;
  if (i < n) out[i] = (_Float16)in[i];
}

// ---------------------------------------------------------------------------
// QKV projection: Y[8192,3072] = Xh @ Wqkv^T + b, scattered into Q/K/Vt.
// Block = 4 waves; wave computes a 64x64 tile (16 accum frags = 128 VGPRs);
// per k-step: 16 B128 loads : 16 WMMAs. Grid (8192/256, 3072/64).
__global__ __launch_bounds__(128) void qkv_gemm(
    const _Float16* __restrict__ X, const _Float16* __restrict__ W,
    const float* __restrict__ bias, _Float16* __restrict__ Q,
    _Float16* __restrict__ K, _Float16* __restrict__ Vt) {
  const int w = threadIdx.x >> 5;
  const int lane = threadIdx.x & 31;
  const int hi = lane >> 4, lo = lane & 15;
  const int m0 = blockIdx.x * 256 + w * 64;
  const int n0 = blockIdx.y * 64;

  v8f acc[4][4] = {};
  for (int k0 = 0; k0 < 1024; k0 += 32) {
    if (k0 + 32 < 1024) {
      __builtin_prefetch(X + (size_t)m0 * 1024 + k0 + 32, 0, 1);
      __builtin_prefetch(W + (size_t)n0 * 1024 + k0 + 32, 0, 1);
    }
    v16h a[4];
#pragma unroll
    for (int i = 0; i < 4; ++i) a[i] = load_a_frag(X, 1024, m0 + 16 * i, k0);
#pragma unroll
    for (int j = 0; j < 4; ++j) {
      v16h bf = load_b_frag(W, 1024, n0 + 16 * j, k0);
#pragma unroll
      for (int i = 0; i < 4; ++i) acc[i][j] = wmma_f16(a[i], bf, acc[i][j]);
    }
  }
  // Epilogue: add bias, scatter to [B,H,N,D] (q,k) / [B,H,D,N] (v).
#pragma unroll
  for (int i = 0; i < 4; ++i) {
#pragma unroll
    for (int j = 0; j < 4; ++j) {
      const int n = n0 + 16 * j + lo;          // column in [0,3072)
      const int which = n >> 10;               // 0=q 1=k 2=v
      const int rem = n & 1023;
      const int h = rem >> 6, d = rem & 63;
      const float bv = bias[n];
#pragma unroll
      for (int r = 0; r < 8; ++r) {
        const int m = m0 + 16 * i + 8 * hi + r;  // row in [0,8192)
        const int b = m >> 11, pos = m & 2047;
        const _Float16 val = (_Float16)(acc[i][j][r] + bv);
        const size_t bh = (size_t)(b * 16 + h);
        if (which == 0)      Q[(bh * 2048 + pos) * 64 + d] = val;
        else if (which == 1) K[(bh * 2048 + pos) * 64 + d] = val;
        else                 Vt[(bh * 64 + d) * 2048 + pos] = val;
      }
    }
  }
}

// ---------------------------------------------------------------------------
// RoPE in-place on Q and K; each thread rotates the (d, d+32) pair.
// cos/sin identical for d and d+32 (emb = concat(freqs,freqs)).
// Attention scale 1/sqrt(64)=0.125 folded into Q.
__global__ void rope_kernel(_Float16* __restrict__ Q, _Float16* __restrict__ K) {
  const int tid = blockIdx.x * blockDim.x + threadIdx.x;  // B*H*N*32 = 2^22
  const int j = tid & 31;
  const int pos = (tid >> 5) & 2047;
  const int bh = tid >> 16;
  const size_t base = ((size_t)bh * 2048 + pos) * 64;
  const float inv_freq = __powf(10000.0f, -(float)j * (1.0f / 32.0f));
  const float ang = (float)pos * inv_freq;
  float s, c;
  __sincosf(ang, &s, &c);
  float q1 = (float)Q[base + j], q2 = (float)Q[base + j + 32];
  Q[base + j]      = (_Float16)((q1 * c - q2 * s) * 0.125f);
  Q[base + j + 32] = (_Float16)((q2 * c + q1 * s) * 0.125f);
  float k1 = (float)K[base + j], k2 = (float)K[base + j + 32];
  K[base + j]      = (_Float16)(k1 * c - k2 * s);
  K[base + j + 32] = (_Float16)(k2 * c + k1 * s);
}

// ---------------------------------------------------------------------------
// Flash attention. 1 wave = 32 q-rows (two 16-row tiles) of one (b,h).
// Computes S^T = K @ Q^T so the S^T C-fragment (lane = q-row i, vgpr r =
// kv-row j offset) is *exactly* the A-fragment layout of P for O += P @ V:
// no shuffles, no LDS transpose. K a-frags and V b-frags are shared across
// both q-tiles -> per 32-KV step: 16 B128 loads : 16 WMMAs.
// Softmax stats are per-lane (i = lane&15), halves combined via shfl_xor(16).
__global__ __launch_bounds__(128) void flash_attn(
    const _Float16* __restrict__ Q, const _Float16* __restrict__ K,
    const _Float16* __restrict__ Vt, _Float16* __restrict__ O /*[B,N,C]*/) {
  const int w = threadIdx.x >> 5;
  const int lane = threadIdx.x & 31;
  const int hi = lane >> 4, lo = lane & 15;
  const int g = blockIdx.x * 4 + w;         // global wave id
  const int bh = g >> 6;                    // 64 row-blocks (32 rows) per (b,h)
  const int i0 = (g & 63) << 5;
  const int b = bh >> 4, h = bh & 15;
  const _Float16* Qbh = Q + (size_t)bh * 2048 * 64;
  const _Float16* Kbh = K + (size_t)bh * 2048 * 64;
  const _Float16* Vbh = Vt + (size_t)bh * 64 * 2048;

  // Loop-invariant Q B-fragments: two q-tiles x two d-halves.
  v16h bq[2][2];
#pragma unroll
  for (int q = 0; q < 2; ++q) {
    bq[q][0] = load_b_frag(Qbh, 64, i0 + 16 * q, 0);
    bq[q][1] = load_b_frag(Qbh, 64, i0 + 16 * q, 32);
  }

  v8f o[2][4] = {};                         // per q-tile: 16 x 64 O accum
  float m[2] = {-3.0e38f, -3.0e38f};        // per-lane stats for q-row i=lane&15
  float l[2] = {0.0f, 0.0f};

  for (int j0 = 0; j0 < 2048; j0 += 32) {
    // Shared K a-frags (two j-tiles x two d-halves) and V b-frags (4 d-tiles).
    const v16h ak00 = load_a_frag(Kbh, 64, j0, 0);
    const v16h ak01 = load_a_frag(Kbh, 64, j0, 32);
    const v16h ak10 = load_a_frag(Kbh, 64, j0 + 16, 0);
    const v16h ak11 = load_a_frag(Kbh, 64, j0 + 16, 32);
    v16h bv[4];
#pragma unroll
    for (int t = 0; t < 4; ++t) bv[t] = load_b_frag(Vbh, 2048, t * 16, j0);

#pragma unroll
    for (int q = 0; q < 2; ++q) {
      // S^T tiles: (j0..j0+15) x qtile and (j0+16..j0+31) x qtile
      v8f s0 = {}, s1 = {};
      s0 = wmma_f16(ak00, bq[q][0], s0);
      s0 = wmma_f16(ak01, bq[q][1], s0);
      s1 = wmma_f16(ak10, bq[q][0], s1);
      s1 = wmma_f16(ak11, bq[q][1], s1);

      // Online softmax (scale already folded into Q).
      float smax = -3.0e38f;
#pragma unroll
      for (int r = 0; r < 8; ++r) smax = fmaxf(smax, fmaxf(s0[r], s1[r]));
      smax = fmaxf(smax, __shfl_xor(smax, 16, 32));
      const float mnew = fmaxf(m[q], smax);
      const float alpha = __expf(m[q] - mnew);
      float lsum = 0.0f;
      v16h ap;  // P as A-frag: e<8 -> j=8*hi+e (tile0), e>=8 -> 16+8*hi+(e-8)
#pragma unroll
      for (int r = 0; r < 8; ++r) {
        const float p0 = __expf(s0[r] - mnew);
        const float p1 = __expf(s1[r] - mnew);
        lsum += p0 + p1;
        ap[r] = (_Float16)p0;
        ap[r + 8] = (_Float16)p1;
      }
      lsum += __shfl_xor(lsum, 16, 32);
      l[q] = l[q] * alpha + lsum;
      m[q] = mnew;

      // Rescale O: O's rows are i = r+8*hi, so gather alpha per row.
      float af[8];
#pragma unroll
      for (int r = 0; r < 8; ++r) af[r] = __shfl(alpha, 8 * hi + r, 32);
#pragma unroll
      for (int t = 0; t < 4; ++t)
#pragma unroll
        for (int r = 0; r < 8; ++r) o[q][t][r] *= af[r];

      // O += P @ V ; B-fragment straight from Vt [D,N] row-major.
#pragma unroll
      for (int t = 0; t < 4; ++t) o[q][t] = wmma_f16(ap, bv[t], o[q][t]);
    }
  }

  // Normalize and store to Oh[b, pos, h*64 + d]  (row-major [B*N, C]).
#pragma unroll
  for (int q = 0; q < 2; ++q) {
    float lf[8];
#pragma unroll
    for (int r = 0; r < 8; ++r) lf[r] = 1.0f / __shfl(l[q], 8 * hi + r, 32);
#pragma unroll
    for (int t = 0; t < 4; ++t)
#pragma unroll
      for (int r = 0; r < 8; ++r) {
        const int pos = i0 + 16 * q + 8 * hi + r;
        const size_t idx =
            ((size_t)b * 2048 + pos) * 1024 + h * 64 + t * 16 + lo;
        O[idx] = (_Float16)(o[q][t][r] * lf[r]);
      }
  }
}

// ---------------------------------------------------------------------------
// Output projection: out[8192,1024] (f32) = Oh @ Wout^T + b_out.
// Same 64x64-per-wave tiling as qkv_gemm. Grid (8192/256, 1024/64).
__global__ __launch_bounds__(128) void out_gemm(
    const _Float16* __restrict__ X, const _Float16* __restrict__ W,
    const float* __restrict__ bias, float* __restrict__ Y) {
  const int w = threadIdx.x >> 5;
  const int lane = threadIdx.x & 31;
  const int hi = lane >> 4, lo = lane & 15;
  const int m0 = blockIdx.x * 256 + w * 64;
  const int n0 = blockIdx.y * 64;

  v8f acc[4][4] = {};
  for (int k0 = 0; k0 < 1024; k0 += 32) {
    if (k0 + 32 < 1024) {
      __builtin_prefetch(X + (size_t)m0 * 1024 + k0 + 32, 0, 1);
      __builtin_prefetch(W + (size_t)n0 * 1024 + k0 + 32, 0, 1);
    }
    v16h a[4];
#pragma unroll
    for (int i = 0; i < 4; ++i) a[i] = load_a_frag(X, 1024, m0 + 16 * i, k0);
#pragma unroll
    for (int j = 0; j < 4; ++j) {
      v16h bf = load_b_frag(W, 1024, n0 + 16 * j, k0);
#pragma unroll
      for (int i = 0; i < 4; ++i) acc[i][j] = wmma_f16(a[i], bf, acc[i][j]);
    }
  }
#pragma unroll
  for (int i = 0; i < 4; ++i)
#pragma unroll
    for (int j = 0; j < 4; ++j) {
      const int n = n0 + 16 * j + lo;
      const float bv = bias[n];
#pragma unroll
      for (int r = 0; r < 8; ++r) {
        const int mrow = m0 + 16 * i + 8 * hi + r;
        Y[(size_t)mrow * 1024 + n] = acc[i][j][r] + bv;
      }
    }
}

// ---------------------------------------------------------------------------
extern "C" void kernel_launch(void* const* d_in, const int* in_sizes, int n_in,
                              void* d_out, int out_size, void* d_ws, size_t ws_size,
                              hipStream_t stream) {
  const float* x     = (const float*)d_in[0];
  const float* W_qkv = (const float*)d_in[1];
  const float* b_qkv = (const float*)d_in[2];
  const float* W_out = (const float*)d_in[3];
  const float* b_out = (const float*)d_in[4];
  float* out = (float*)d_out;

  _Float16* ws = (_Float16*)d_ws;
  const size_t nX = (size_t)8192 * 1024;          // 8 Mi halves
  const size_t nWq = (size_t)3072 * 1024;
  const size_t nWo = (size_t)1024 * 1024;
  const size_t nHd = (size_t)4 * 16 * 2048 * 64;  // B*H*N*D
  _Float16* Xh  = ws;            // later reused as attention output Oh
  _Float16* Wqh = Xh + nX;
  _Float16* Woh = Wqh + nWq;
  _Float16* Qh  = Woh + nWo;
  _Float16* Kh  = Qh + nHd;
  _Float16* Vt  = Kh + nHd;
  // total: nX+nWq+nWo+3*nHd = 37,748,736 halves = 75,497,472 bytes

  cvt_f32_f16<<<(int)((nX + 255) / 256), 256, 0, stream>>>(x, Xh, (int)nX);
  cvt_f32_f16<<<(int)((nWq + 255) / 256), 256, 0, stream>>>(W_qkv, Wqh, (int)nWq);
  cvt_f32_f16<<<(int)((nWo + 255) / 256), 256, 0, stream>>>(W_out, Woh, (int)nWo);

  qkv_gemm<<<dim3(32, 48), 128, 0, stream>>>(Xh, Wqh, b_qkv, Qh, Kh, Vt);
  rope_kernel<<<16384, 256, 0, stream>>>(Qh, Kh);
  flash_attn<<<1024, 128, 0, stream>>>(Qh, Kh, Vt, Xh);   // Oh aliases Xh
  out_gemm<<<dim3(32, 16), 128, 0, stream>>>(Xh, Woh, b_out, out);
}
